// GraphEncoderGNN_10917806866970
// MI455X (gfx1250) — compile-verified, matrix operand order checked by reading, not run on hardware
//
#include <hip/hip_runtime.h>
#include <hip/hip_bf16.h>

#define N_NODES  50000
#define N_EDGES  800000
#define D_IN     96
#define D_HID    256
#define N_HID    512
#define N_OUT    768
#define N_GRAPHS 256
#define K_TOT    192   // 96 (agg @ W_rel) + 96 (x @ W_root) fused

typedef __attribute__((ext_vector_type(16))) __bf16 v16bf;
typedef __attribute__((ext_vector_type(8)))  float  v8f;
typedef __attribute__((ext_vector_type(4)))  int    v4i;

#define GLOBAL_AS __attribute__((address_space(1)))
#define LDS_AS    __attribute__((address_space(3)))

#if __has_builtin(__builtin_amdgcn_global_load_async_to_lds_b128)
#define HAS_ASYNC_LDS 1
#else
#define HAS_ASYNC_LDS 0
#endif

__device__ __forceinline__ void wait_asynccnt0() {
#if __has_builtin(__builtin_amdgcn_s_wait_asynccnt)
  __builtin_amdgcn_s_wait_asynccnt(0);
#else
  asm volatile("s_wait_asynccnt 0x0" ::: "memory");
#endif
}

// fp32 -> bf16: let the backend use the native convert instruction.
__device__ __forceinline__ __bf16 f2bf(float f) { return (__bf16)f; }

// Pack 8 contiguous fp32 (32B, 16B-aligned) into frag[base..base+7] as bf16.
__device__ __forceinline__ void pack8_f32(v16bf& frag, int base, const float* p) {
  float4 lo = *(const float4*)p;
  float4 hi = *(const float4*)(p + 4);
  frag[base + 0] = f2bf(lo.x); frag[base + 1] = f2bf(lo.y);
  frag[base + 2] = f2bf(lo.z); frag[base + 3] = f2bf(lo.w);
  frag[base + 4] = f2bf(hi.x); frag[base + 5] = f2bf(hi.y);
  frag[base + 6] = f2bf(hi.z); frag[base + 7] = f2bf(hi.w);
}

// Pack 8 contiguous bf16 (16B, aligned) from LDS into frag[base..base+7].
__device__ __forceinline__ void pack8_lds(v16bf& frag, int base, const __bf16* p) {
  union { uint4 u; __bf16 b[8]; } cvt;
  cvt.u = *(const uint4*)p;
  #pragma unroll
  for (int i = 0; i < 8; ++i) frag[base + i] = cvt.b[i];
}

// ---------------- kernel 1: zero agg + pooled-max buffer ----------------
__global__ void __launch_bounds__(256) init_kernel(float* __restrict__ agg,
                                                   float* __restrict__ g) {
  size_t i = (size_t)blockIdx.x * blockDim.x + threadIdx.x;
  size_t step = (size_t)gridDim.x * blockDim.x;
  for (size_t idx = i; idx < (size_t)N_NODES * D_IN; idx += step) agg[idx] = 0.f;
  for (size_t idx = i; idx < (size_t)N_GRAPHS * D_HID; idx += step) g[idx] = 0.f;
}

// ---------------- kernel 2: edge scatter-add (segment_sum) ----------------
// One wave32 per edge: lane l handles features l, l+32, l+64. agg (19MB) and
// x (19MB) are L2-resident (192MB L2), so atomics run at L2 throughput.
__global__ void __launch_bounds__(256) scatter_add_kernel(
    const float* __restrict__ x, const int* __restrict__ src,
    const int* __restrict__ dst, float* __restrict__ agg) {
  int wave = (int)(((size_t)blockIdx.x * blockDim.x + threadIdx.x) >> 5);
  int lane = threadIdx.x & 31;
  if (wave >= N_EDGES) return;
  int s = src[wave];
  int d = dst[wave];
  const float* xr = x + (size_t)s * D_IN;
  float* ar = agg + (size_t)d * D_IN;
  #pragma unroll
  for (int i = 0; i < 3; ++i) {
    int f = lane + i * 32;
    atomicAdd(&ar[f], xr[f]);
  }
}

// ---- kernel 3: fused [agg|x] @ [W_rel;W_root] + b, ReLU, segment-max pool ----
// Block = 256 threads (8 waves). Block tile: 128 rows x 128 cols (grid.y picks
// col half). Weights are staged into LDS (transposed, bf16) via async
// global->LDS copies (ASYNCcnt path) through an fp32 staging buffer.
// h is never written to memory: epilogue does bias+relu and uint atomicMax
// pooling (valid since h >= 0 after relu).
__global__ void __launch_bounds__(256) gemm_pool_kernel(
    const float* __restrict__ agg, const float* __restrict__ x,
    const float* __restrict__ W_rel, const float* __restrict__ W_root,
    const float* __restrict__ b_rel, const int* __restrict__ batch,
    unsigned* __restrict__ g) {
  __shared__ __bf16 sWT[128 * K_TOT];                 // [local col][k], 48 KB
  __shared__ __align__(16) float sStage[16 * 128];    // 8 KB fp32 staging
  const int n0 = blockIdx.y * 128;

  // Stage weights in 12 chunks of 16 k-rows x 128 cols:
  // chunks 0..5 from W_rel (k 0..95), 6..11 from W_root (k 96..191).
  for (int c = 0; c < 12; ++c) {
    const int half = c / 6;
    const int kb = (c - half * 6) * 16;
    const float* Wsrc = half ? W_root : W_rel;
#if HAS_ASYNC_LDS
    #pragma unroll
    for (int it = 0; it < 2; ++it) {                  // 2 x 256 lanes x 16B = 8KB
      int elt = (it * 256 + threadIdx.x) * 4;         // float index in chunk
      int kk = elt >> 7, nn = elt & 127;
      const float* gsrc = Wsrc + (size_t)(kb + kk) * D_HID + n0 + nn;
      __builtin_amdgcn_global_load_async_to_lds_b128(
          (GLOBAL_AS v4i*)gsrc, (LDS_AS v4i*)&sStage[elt], 0, 0);
    }
    wait_asynccnt0();
    __syncthreads();
#else
    #pragma unroll
    for (int it = 0; it < 2; ++it) {
      int elt = (it * 256 + threadIdx.x) * 4;
      int kk = elt >> 7, nn = elt & 127;
      *(float4*)&sStage[elt] =
          *(const float4*)(Wsrc + (size_t)(kb + kk) * D_HID + n0 + nn);
    }
    __syncthreads();
#endif
    // Convert fp32 staging -> transposed bf16 tile.
    for (int i = threadIdx.x; i < 16 * 128; i += 256) {
      int kk = i >> 7, nn = i & 127;
      sWT[nn * K_TOT + half * 96 + kb + kk] = f2bf(sStage[i]);
    }
    __syncthreads();
  }

  const int wave = threadIdx.x >> 5;
  const int lane = threadIdx.x & 31;
  const int hi = lane >> 4, lo = lane & 15;
  const int m0 = blockIdx.x * 128 + wave * 16;

  const int arow  = m0 + lo;
  const int arowc = arow < N_NODES ? arow : N_NODES - 1;  // clamp for loads
  const float* aggr = agg + (size_t)arowc * D_IN;
  const float* xr   = x   + (size_t)arowc * D_IN;

  v8f acc[8];
  #pragma unroll
  for (int t = 0; t < 8; ++t) acc[t] = (v8f){0, 0, 0, 0, 0, 0, 0, 0};

  #pragma unroll
  for (int kc = 0; kc < 6; ++kc) {
    const int k0 = kc * 32;
    // ISA 16-bit A 16x32 layout: lane holds K = 8*hi + (j<8 ? j : j+8) + k0
    // -> two contiguous 8-float runs at r0 and r0+16 (both in same source).
    const int r0 = k0 + hi * 8;
    const float* srow = (r0 < D_IN) ? aggr : xr;
    const int rk = (r0 < D_IN) ? r0 : r0 - D_IN;
    v16bf afrag;
    pack8_f32(afrag, 0, srow + rk);
    pack8_f32(afrag, 8, srow + rk + 16);

    #pragma unroll
    for (int ct = 0; ct < 8; ++ct) {
      const __bf16* wp = &sWT[(ct * 16 + lo) * K_TOT + r0];
      v16bf bfrag;
      pack8_lds(bfrag, 0, wp);
      pack8_lds(bfrag, 8, wp + 16);
      acc[ct] = __builtin_amdgcn_wmma_f32_16x16x32_bf16(
          false, afrag, false, bfrag, (short)0, acc[ct], false, false);
    }
  }

  // Epilogue: bias + ReLU + segment-max into g (uint max valid for vals >= 0).
  int gid[8];
  #pragma unroll
  for (int r = 0; r < 8; ++r) {
    int row = m0 + hi * 8 + r;  // C layout: M = r + 8*hi
    gid[r] = (row < N_NODES) ? batch[row] : -1;
  }
  #pragma unroll
  for (int ct = 0; ct < 8; ++ct) {
    const int col = n0 + ct * 16 + lo;
    const float bias = b_rel[col];
    #pragma unroll
    for (int r = 0; r < 8; ++r) {
      if (gid[r] >= 0) {
        float v = acc[ct][r] + bias;
        v = v > 0.f ? v : 0.f;
        atomicMax(&g[gid[r] * D_HID + col], __builtin_bit_cast(unsigned, v));
      }
    }
  }
}

// ---------------- kernel 4: g2 = relu(g @ W1 + b1), 256x256x512 ----------------
__global__ void __launch_bounds__(256) mlp1_kernel(
    const float* __restrict__ g, const float* __restrict__ W1,
    const float* __restrict__ b1, float* __restrict__ g2) {
  const int wave = threadIdx.x >> 5, lane = threadIdx.x & 31;
  const int hi = lane >> 4, lo = lane & 15;
  const int w = blockIdx.x * 8 + wave;       // 512 tiles = 16 x 32
  const int mt = w >> 5, nt = w & 31;
  const int row = mt * 16 + lo;
  const int col = nt * 16 + lo;

  v8f acc = (v8f){0, 0, 0, 0, 0, 0, 0, 0};
  #pragma unroll
  for (int kc = 0; kc < 8; ++kc) {
    const int r0 = kc * 32 + hi * 8;
    v16bf afrag, bfrag;
    pack8_f32(afrag, 0, g + row * D_HID + r0);
    pack8_f32(afrag, 8, g + row * D_HID + r0 + 16);
    #pragma unroll
    for (int j = 0; j < 8; ++j) {
      bfrag[j]     = f2bf(W1[(r0 + j)      * N_HID + col]);
      bfrag[j + 8] = f2bf(W1[(r0 + 16 + j) * N_HID + col]);
    }
    acc = __builtin_amdgcn_wmma_f32_16x16x32_bf16(
        false, afrag, false, bfrag, (short)0, acc, false, false);
  }
  const float bias = b1[col];
  #pragma unroll
  for (int r = 0; r < 8; ++r) {
    const int orow = mt * 16 + hi * 8 + r;
    float v = acc[r] + bias;
    g2[orow * N_HID + col] = v > 0.f ? v : 0.f;
  }
}

// ---------------- kernel 5: out = g2 @ W2 + b2, 256x512x768 ----------------
__global__ void __launch_bounds__(256) mlp2_kernel(
    const float* __restrict__ g2, const float* __restrict__ W2,
    const float* __restrict__ b2, float* __restrict__ out) {
  const int wave = threadIdx.x >> 5, lane = threadIdx.x & 31;
  const int hi = lane >> 4, lo = lane & 15;
  const int w = blockIdx.x * 8 + wave;       // 768 tiles = 16 x 48
  const int mt = w / 48, nt = w % 48;
  const int row = mt * 16 + lo;
  const int col = nt * 16 + lo;

  v8f acc = (v8f){0, 0, 0, 0, 0, 0, 0, 0};
  #pragma unroll
  for (int kc = 0; kc < 16; ++kc) {
    const int r0 = kc * 32 + hi * 8;
    v16bf afrag, bfrag;
    pack8_f32(afrag, 0, g2 + row * N_HID + r0);
    pack8_f32(afrag, 8, g2 + row * N_HID + r0 + 16);
    #pragma unroll
    for (int j = 0; j < 8; ++j) {
      bfrag[j]     = f2bf(W2[(r0 + j)      * N_OUT + col]);
      bfrag[j + 8] = f2bf(W2[(r0 + 16 + j) * N_OUT + col]);
    }
    acc = __builtin_amdgcn_wmma_f32_16x16x32_bf16(
        false, afrag, false, bfrag, (short)0, acc, false, false);
  }
  const float bias = b2[col];
  #pragma unroll
  for (int r = 0; r < 8; ++r) {
    const int orow = mt * 16 + hi * 8 + r;
    out[orow * N_OUT + col] = acc[r] + bias;
  }
}

extern "C" void kernel_launch(void* const* d_in, const int* in_sizes, int n_in,
                              void* d_out, int out_size, void* d_ws, size_t ws_size,
                              hipStream_t stream) {
  (void)in_sizes; (void)n_in; (void)out_size; (void)ws_size;
  const float* x      = (const float*)d_in[0];
  const int*   eidx   = (const int*)  d_in[1];   // [2, N_EDGES]: src row then dst row
  const int*   batch  = (const int*)  d_in[2];
  const float* W_rel  = (const float*)d_in[3];
  const float* b_rel  = (const float*)d_in[4];
  const float* W_root = (const float*)d_in[5];
  const float* W1     = (const float*)d_in[6];
  const float* b1     = (const float*)d_in[7];
  const float* W2     = (const float*)d_in[8];
  const float* b2     = (const float*)d_in[9];
  float* out = (float*)d_out;

  // Workspace: agg [50000*96] | g [256*256] | g2 [256*512]  (~20.3 MB)
  float* agg = (float*)d_ws;
  float* g   = agg + (size_t)N_NODES * D_IN;
  float* g2  = g + (size_t)N_GRAPHS * D_HID;

  init_kernel<<<1024, 256, 0, stream>>>(agg, g);
  scatter_add_kernel<<<(N_EDGES * 32 + 255) / 256, 256, 0, stream>>>(
      x, eidx, eidx + N_EDGES, agg);
  dim3 grid3((N_NODES + 127) / 128, 2);
  gemm_pool_kernel<<<grid3, 256, 0, stream>>>(agg, x, W_rel, W_root, b_rel,
                                              batch, (unsigned*)g);
  mlp1_kernel<<<64, 256, 0, stream>>>(g, W1, b1, g2);
  mlp2_kernel<<<96, 256, 0, stream>>>(g2, W2, b2, out);
}
